// PagedAttention_33878702031091
// MI455X (gfx1250) — matrix operand order repaired and checked by reading, not compiled, not run on hardware
//
#include <hip/hip_runtime.h>
#include <cstdint>

// ---------------- problem constants ----------------
#define HIDDEN   2048
#define HEADS    16
#define HEAD_DIM 128
#define BATCH    8
#define QLEN     1024
#define KVLEN    4096
#define M_TOT    (BATCH * QLEN)        // 8192
#define ATT_SCALE 0.08838834764831845f // 1/sqrt(128)

typedef __attribute__((ext_vector_type(16))) _Float16 v16h;
typedef __attribute__((ext_vector_type(8)))  _Float16 v8h;
typedef __attribute__((ext_vector_type(8)))  float    v8f;

static __device__ __forceinline__ v8f wmma_f16(v16h a, v16h b, v8f c) {
  // D = A(16x32 f16) * B(32x16 f16) + C(16x16 f32)
  return __builtin_amdgcn_wmma_f32_16x16x32_f16(
      false, a, false, b, (short)0, c, false, false);
}

// -------- CDNA5 async global->LDS copy (ASYNCcnt-tracked), inline asm ------
// VDST VGPR = per-lane LDS byte offset (== low 32 bits of generic pointer,
// since flat LDS addresses are {aperture_hi, as3_offset}); VADDR = 64b global.
static __device__ __forceinline__ void cp_g2l_16h(_Float16* l, const _Float16* g) {
  uint32_t l0 = (uint32_t)(uintptr_t)l;
  asm volatile("global_load_async_to_lds_b128 %0, %1, off"
               :: "v"(l0), "v"((const void*)g) : "memory");
  asm volatile("global_load_async_to_lds_b128 %0, %1, off"
               :: "v"((uint32_t)(l0 + 16)), "v"((const void*)(g + 8)) : "memory");
}

static __device__ __forceinline__ void stage_fence() {
  asm volatile("s_wait_asynccnt 0x0" ::: "memory");
  __syncthreads();   // also waits DScnt for plain ds stores
}

// ---------------- fp32 -> f16 conversion ----------------
__global__ __launch_bounds__(256) void cvt_f32_to_f16(
    const float* __restrict__ src, _Float16* __restrict__ dst, int n) {
  int i = (blockIdx.x * 256 + threadIdx.x) * 8;
  if (i + 8 <= n) {
#pragma unroll
    for (int j = 0; j < 8; ++j) dst[i + j] = (_Float16)src[i + j];
  }
}

// ---- WMMA GEMM: Out[M,N] = A[M,K](f16) @ W[N,K](f16)^T, block-tiled -------
// 256 threads = 8 waves. Block tile 128x128, K-step 32, LDS double-buffered
// via async loads. Wave tile 64x32 = 4x2 WMMA accumulators.
#define GT_M 128
#define GT_N 128
#define GT_K 32
#define GPAD 40 // padded LDS row stride (halves): 20 dwords -> conflict-free

template <bool OUT_F16>
__global__ __launch_bounds__(256) void gemm_wmma2(
    const _Float16* __restrict__ A, const _Float16* __restrict__ Bw,
    void* __restrict__ Out, int M, int N, int K) {
  __shared__ _Float16 ash[2][GT_M * GPAD];
  __shared__ _Float16 bsh[2][GT_N * GPAD];

  const int tid  = threadIdx.x;
  const int lane = tid & 31;
  const int wave = tid >> 5;     // 0..7
  const int wm   = wave & 1;     // 2 along M
  const int wn   = wave >> 1;    // 4 along N
  const int m0 = blockIdx.x * GT_M;
  const int n0 = blockIdx.y * GT_N;
  const int mr = lane & 15;
  const int kg = lane >> 4;

  // staging: 256 threads x 32B = one 128x32 f16 tile
  const int srow = tid >> 1;
  const int scol = (tid & 1) * 16;
  const int soff = srow * GPAD + scol;
  const _Float16* ag = A  + (size_t)(m0 + srow) * K + scol;
  const _Float16* bg = Bw + (size_t)(n0 + srow) * K + scol;

  v8f acc[4][2];
#pragma unroll
  for (int i = 0; i < 4; ++i)
#pragma unroll
    for (int j = 0; j < 2; ++j) acc[i][j] = (v8f){};

  cp_g2l_16h(&ash[0][soff], ag);
  cp_g2l_16h(&bsh[0][soff], bg);
  stage_fence();

  int buf = 0;
  for (int k = 0; k < K; k += GT_K) {
    int kn = k + GT_K;
    if (kn < K) {                       // async-prefetch next K-slab
      cp_g2l_16h(&ash[buf ^ 1][soff], ag + kn);
      cp_g2l_16h(&bsh[buf ^ 1][soff], bg + kn);
    }

    const _Float16* ab = &ash[buf][(wm * 64) * GPAD];
    const _Float16* bb = &bsh[buf][(wn * 32) * GPAD];
    v16h af[4], bf[2];
#pragma unroll
    for (int i = 0; i < 4; ++i) {
      const _Float16* ar = ab + (i * 16 + mr) * GPAD;
#pragma unroll
      for (int e = 0; e < 8; ++e) {
        af[i][e]     = ar[kg * 16 + e];
        af[i][8 + e] = ar[kg * 16 + 8 + e];
      }
    }
#pragma unroll
    for (int j = 0; j < 2; ++j) {
      const _Float16* br = bb + (j * 16 + mr) * GPAD + kg * 16;
#pragma unroll
      for (int e = 0; e < 16; ++e) bf[j][e] = br[e];
    }
#pragma unroll
    for (int i = 0; i < 4; ++i)
#pragma unroll
      for (int j = 0; j < 2; ++j)
        acc[i][j] = wmma_f16(af[i], bf[j], acc[i][j]);

    stage_fence();
    buf ^= 1;
  }

#pragma unroll
  for (int i = 0; i < 4; ++i)
#pragma unroll
    for (int j = 0; j < 2; ++j) {
      const int orow = m0 + wm * 64 + i * 16 + kg * 8;
      const int ocol = n0 + wn * 32 + j * 16 + mr;
      if constexpr (OUT_F16) {
        _Float16* o = (_Float16*)Out;
#pragma unroll
        for (int r = 0; r < 8; ++r)
          o[(size_t)(orow + r) * N + ocol] = (_Float16)acc[i][j][r];
      } else {
        float* o = (float*)Out;
#pragma unroll
        for (int r = 0; r < 8; ++r)
          o[(size_t)(orow + r) * N + ocol] = acc[i][j][r];
      }
    }
}

// ------------- flash attention: block = 4 waves sharing one (b,h) ----------
// Each wave owns one 16-row q-tile; K/V chunks (32 kv rows) are staged ONCE
// per block: K via async global->LDS (double-buffered), V transposed
// cooperatively. 4x global-traffic reduction vs wave-private loads.
#define ATT_WAVES 4
#define KPAD    136  // [kv][d] row stride (halves): 68 dwords -> stride 4 banks
#define VSTRIDE 34   // [d][kv] transposed row stride (halves)
#define PSTRIDE 40   // P staging row stride (halves)

__global__ __launch_bounds__(ATT_WAVES * 32) void attn_wmma(
    const _Float16* __restrict__ Qh, const _Float16* __restrict__ Kh,
    const _Float16* __restrict__ Vh, _Float16* __restrict__ Oh) {
  __shared__ _Float16 ksh[2][32 * KPAD];             // K chunk  [kv][d]
  __shared__ _Float16 vsh[2][HEAD_DIM * VSTRIDE];    // V^T chunk [d][kv]
  __shared__ _Float16 pbuf[ATT_WAVES][16 * PSTRIDE]; // per-wave P tile

  const int tid  = threadIdx.x;
  const int lane = tid & 31;
  const int wave = tid >> 5;
  // block -> (b, h, q-group of 4 tiles)
  const int bq = blockIdx.x & 15;
  const int h  = (blockIdx.x >> 4) & 15;
  const int b  = blockIdx.x >> 8;
  const int qt = bq * 4 + wave;
  const int mr = lane & 15;
  const int kg = lane >> 4;

  // cooperative staging coords
  const int krow = tid >> 2, kcol = (tid & 3) * 32;  // K: 64B per thread
  const int vrow = tid & 31, vseg = (tid >> 5) * 32; // V: quarter-row per thread

  // per-wave Q fragments (4 K-steps over D=128)
  v16h qa[4];
  const _Float16* qrow =
      Qh + ((size_t)(b * QLEN + qt * 16 + mr)) * HIDDEN + h * HEAD_DIM;
#pragma unroll
  for (int ks = 0; ks < 4; ++ks)
#pragma unroll
    for (int i = 0; i < 8; ++i) {
      qa[ks][i]     = qrow[ks * 32 + kg * 16 + i];
      qa[ks][8 + i] = qrow[ks * 32 + kg * 16 + 8 + i];
    }

  v8f o[8];
#pragma unroll
  for (int t = 0; t < 8; ++t) o[t] = (v8f){};
  float mrow[8], lrow[8];
#pragma unroll
  for (int r = 0; r < 8; ++r) { mrow[r] = -1e30f; lrow[r] = 0.0f; }

  _Float16* pb = pbuf[wave];
  const _Float16* kbase = Kh + (size_t)h * HEAD_DIM;
  const _Float16* vbase = Vh + (size_t)h * HEAD_DIM;

  // ---- stage chunk 0 ----
  cp_g2l_16h(&ksh[0][krow * KPAD + kcol],      kbase + (size_t)krow * HIDDEN + kcol);
  cp_g2l_16h(&ksh[0][krow * KPAD + kcol + 16], kbase + (size_t)krow * HIDDEN + kcol + 16);
  {
    const _Float16* vr = vbase + (size_t)vrow * HIDDEN + vseg;
#pragma unroll
    for (int d = 0; d < 32; ++d) vsh[0][(vseg + d) * VSTRIDE + vrow] = vr[d];
  }
  stage_fence();

  int buf = 0;
  for (int kb = 0; kb < KVLEN; kb += 32) {
    // ---- async-stage next chunk into the other buffer ----
    if (kb + 32 < KVLEN) {
      const _Float16* kn = kbase + (size_t)(kb + 32 + krow) * HIDDEN + kcol;
      cp_g2l_16h(&ksh[buf ^ 1][krow * KPAD + kcol],      kn);
      cp_g2l_16h(&ksh[buf ^ 1][krow * KPAD + kcol + 16], kn + 16);
      const _Float16* vr = vbase + (size_t)(kb + 32 + vrow) * HIDDEN + vseg;
#pragma unroll
      for (int d = 0; d < 32; ++d)
        vsh[buf ^ 1][(vseg + d) * VSTRIDE + vrow] = vr[d];
    }

    const _Float16* kshb = ksh[buf];
    const _Float16* vshb = vsh[buf];

    // ---- S = Q @ K^T: all 8 B-fragments from LDS, then 8 WMMAs ----
    v16h kf[8];
#pragma unroll
    for (int ks = 0; ks < 4; ++ks) {
      const _Float16* k0 = kshb + (size_t)mr * KPAD + ks * 32 + kg * 16;
#pragma unroll
      for (int j = 0; j < 16; ++j) {
        kf[ks * 2 + 0][j] = k0[j];
        kf[ks * 2 + 1][j] = k0[16 * KPAD + j];
      }
    }
    v8f s0 = {}, s1 = {};
#pragma unroll
    for (int ks = 0; ks < 4; ++ks) {
      s0 = wmma_f16(qa[ks], kf[ks * 2 + 0], s0);
      s1 = wmma_f16(qa[ks], kf[ks * 2 + 1], s1);
    }

    // ---- online softmax; write P (f16) into LDS for C->A relayout ----
#pragma unroll
    for (int r = 0; r < 8; ++r) {
      float v0 = s0[r] * ATT_SCALE, v1 = s1[r] * ATT_SCALE;
      float mx = fmaxf(v0, v1);
      mx = fmaxf(mx, __shfl_xor(mx, 1, 32));
      mx = fmaxf(mx, __shfl_xor(mx, 2, 32));
      mx = fmaxf(mx, __shfl_xor(mx, 4, 32));
      mx = fmaxf(mx, __shfl_xor(mx, 8, 32));
      float mnew = fmaxf(mrow[r], mx);
      float corr = __expf(mrow[r] - mnew);
      float p0 = __expf(v0 - mnew);
      float p1 = __expf(v1 - mnew);
      float rs = p0 + p1;
      rs += __shfl_xor(rs, 1, 32);
      rs += __shfl_xor(rs, 2, 32);
      rs += __shfl_xor(rs, 4, 32);
      rs += __shfl_xor(rs, 8, 32);
      lrow[r] = lrow[r] * corr + rs;
      mrow[r] = mnew;
#pragma unroll
      for (int t = 0; t < 8; ++t) o[t][r] *= corr;
      pb[(kg * 8 + r) * PSTRIDE + mr]      = (_Float16)p0;
      pb[(kg * 8 + r) * PSTRIDE + 16 + mr] = (_Float16)p1;
    }

    // ---- reload P as A-fragment (wave-local LDS round trip) ----
    v16h pa;
    {
      const _Float16* pr = pb + mr * PSTRIDE;
#pragma unroll
      for (int i = 0; i < 8; ++i) {
        pa[i]     = pr[kg * 16 + i];
        pa[8 + i] = pr[kg * 16 + 8 + i];
      }
    }

    // ---- O += P @ V over 8 D-tiles (V^T fragments from LDS) ----
#pragma unroll
    for (int nt = 0; nt < 8; ++nt) {
      v16h bv;
      const _Float16* vv = vshb + (nt * 16 + mr) * VSTRIDE + kg * 16;
#pragma unroll
      for (int j = 0; j < 16; ++j) bv[j] = vv[j];
      o[nt] = wmma_f16(pa, bv, o[nt]);
    }

    stage_fence();   // next chunk landed + all waves done with this one
    buf ^= 1;
  }

  float inv[8];
#pragma unroll
  for (int r = 0; r < 8; ++r) inv[r] = 1.0f / lrow[r];
  _Float16* orow =
      Oh + ((size_t)(b * QLEN + qt * 16 + kg * 8)) * HIDDEN + h * HEAD_DIM;
#pragma unroll
  for (int nt = 0; nt < 8; ++nt)
#pragma unroll
    for (int r = 0; r < 8; ++r)
      orow[(size_t)r * HIDDEN + nt * 16 + mr] = (_Float16)(o[nt][r] * inv[r]);
}

// ---------------- host-side launch ----------------
extern "C" void kernel_launch(void* const* d_in, const int* in_sizes, int n_in,
                              void* d_out, int out_size, void* d_ws, size_t ws_size,
                              hipStream_t stream) {
  const float* query = (const float*)d_in[0];
  const float* kv_k  = (const float*)d_in[1];
  const float* kv_v  = (const float*)d_in[2];
  const float* Wq    = (const float*)d_in[3];
  const float* Wo    = (const float*)d_in[4];
  float* out = (float*)d_out;

  _Float16* Xh  = (_Float16*)d_ws;                       // query f16
  _Float16* Wqh = Xh  + (size_t)M_TOT * HIDDEN;
  _Float16* Woh = Wqh + (size_t)HIDDEN * HIDDEN;
  _Float16* Qh  = Woh + (size_t)HIDDEN * HIDDEN;
  _Float16* Kh  = Qh  + (size_t)M_TOT * HIDDEN;
  _Float16* Vh  = Kh  + (size_t)KVLEN * HIDDEN;
  _Float16* Ah  = Vh  + (size_t)KVLEN * HIDDEN;

  const int nq = M_TOT * HIDDEN;      // 16777216
  const int nw = HIDDEN * HIDDEN;     // 4194304
  const int nk = KVLEN * HIDDEN;      // 8388608
  cvt_f32_to_f16<<<nq / 2048, 256, 0, stream>>>(query, Xh, nq);
  cvt_f32_to_f16<<<nw / 2048, 256, 0, stream>>>(Wq, Wqh, nw);
  cvt_f32_to_f16<<<nw / 2048, 256, 0, stream>>>(Wo, Woh, nw);
  cvt_f32_to_f16<<<nk / 2048, 256, 0, stream>>>(kv_k, Kh, nk);
  cvt_f32_to_f16<<<nk / 2048, 256, 0, stream>>>(kv_v, Vh, nk);

  // Q = query @ Wq^T (f16 out)
  gemm_wmma2<true><<<dim3(M_TOT / GT_M, HIDDEN / GT_N), 256, 0, stream>>>(
      Xh, Wqh, (void*)Qh, M_TOT, HIDDEN, HIDDEN);

  // attention: 2048 blocks x 4 waves, block shares (b,h)
  attn_wmma<<<BATCH * HEADS * (QLEN / 16 / ATT_WAVES), ATT_WAVES * 32, 0, stream>>>(
      Qh, Kh, Vh, Ah);

  // out = attn_out @ Wo^T (f32 out)
  gemm_wmma2<false><<<dim3(M_TOT / GT_M, HIDDEN / GT_N), 256, 0, stream>>>(
      Ah, Woh, (void*)out, M_TOT, HIDDEN, HIDDEN);
}
